// EdgeLevelGNN_37151467111036
// MI455X (gfx1250) — matrix operand order repaired and checked by reading, not compiled, not run on hardware
//
#include <hip/hip_runtime.h>

// ---------------------------------------------------------------------------
// EdgeLevelGNN on MI455X (gfx1250): WMMA GCN + fused edge MLP.
// wave32; v_wmma_f32_16x16x32_f16 with f32 accumulation.
// All WMMA fragments are fed by wide contiguous loads:
//   A: two 16B global loads per lane (activations, row-gather)
//   B: one 32B global load per lane from a prepacked fragment-layout copy
//      of the weights (L2-resident; 192MB L2 >> 64KB weights)
// No LDS staging in the GEMM inner loops at all.
// ---------------------------------------------------------------------------

typedef __attribute__((ext_vector_type(16))) _Float16 v16h;
typedef __attribute__((ext_vector_type(8)))  _Float16 v8h;
typedef __attribute__((ext_vector_type(4)))  _Float16 v4h;
typedef __attribute__((ext_vector_type(8)))  float    v8f;

#define N_NODES 50000

// Build 16-wide A fragment from two contiguous 8-element runs.
// A 16x32 layout (ISA 7.12.2): lane half kh in {0,8}; element e -> K offset
// (e&7) + ((e>>3)<<4) + kh, i.e. runs [kh, kh+7] and [kh+16, kh+23].
static __device__ __forceinline__ v16h frag16(v8h lo, v8h hi) {
  v16h a;
#pragma unroll
  for (int e = 0; e < 8; ++e) { a[e] = lo[e]; a[e + 8] = hi[e]; }
  return a;
}

static __device__ __forceinline__ v16h load_frag_a_global(
    const _Float16* __restrict__ rowp, int k0) {
  v8h lo = *(const v8h*)(rowp + k0);
  v8h hi = *(const v8h*)(rowp + k0 + 16);
  return frag16(lo, hi);
}

// ---------------------------------------------------------------------------
// Prepack weights [K x 128] f32 -> WMMA-B fragment layout f16:
//   out[((chunk*8 + nt)*32 + lane)*16 + e] = W[k*128 + n]
//   with k = chunk*32 + (e>>3)*16 + (lane>>4)*8 + (e&7),  n = nt*16 + (lane&15)
// Each lane's 16 B elements become 32 contiguous bytes.
// ---------------------------------------------------------------------------
__global__ void prepack_b_kernel(const float* __restrict__ W,
                                 _Float16* __restrict__ out, int K) {
  int idx = blockIdx.x * blockDim.x + threadIdx.x;  // K*128 total
  if (idx >= K * 128) return;
  const int e     = idx & 15;
  const int lane  = (idx >> 4) & 31;
  const int nt    = (idx >> 9) & 7;
  const int chunk = idx >> 12;
  const int k = chunk * 32 + ((e >> 3) << 4) + ((lane >> 4) << 3) + (e & 7);
  const int n = nt * 16 + (lane & 15);
  out[idx] = (_Float16)W[k * 128 + n];
}

// ---------------------------------------------------------------------------
// GEMM: C[M x 128] = A[M x K] (f16, row-major) * Bpre (prepacked f16).
// Block 256 = 8 waves; wave w owns rows row0 + w*16 .. +15, all 128 cols.
// No LDS, no barriers: pure global_load_b128 + v_wmma inner loop.
// ---------------------------------------------------------------------------
__global__ __launch_bounds__(256)
void gemm_f16_n128_kernel(const _Float16* __restrict__ A,
                          const _Float16* __restrict__ Bpre,
                          float* __restrict__ C, int M, int K) {
  const int t = threadIdx.x;
  const int lane = t & 31;
  const int w = t >> 5;
  const int row0 = blockIdx.x * 128 + w * 16;

  int mrow = row0 + (lane & 15);
  if (mrow > M - 1) mrow = M - 1;                  // clamp loads; stores guarded
  const _Float16* arow = A + (long long)mrow * K + ((lane >> 4) << 3);
  const _Float16* bl = Bpre + (size_t)lane * 16;   // lane's slot in each frag

  v8f acc[8] = {};

  for (int k0 = 0, chunk = 0; k0 < K; k0 += 32, ++chunk) {
    const v16h af = load_frag_a_global(arow, k0);
    const _Float16* bc = bl + (size_t)chunk * (8 * 32 * 16);
#pragma unroll
    for (int nt = 0; nt < 8; ++nt) {
      const v16h bf = *(const v16h*)(bc + nt * (32 * 16));
      acc[nt] = __builtin_amdgcn_wmma_f32_16x16x32_f16(
          false, af, false, bf, (short)0, acc[nt], false, false);
    }
  }

  // C/D layout: VGPR r -> M = row0 + r (+8 for upper lane half), N = lane&15.
#pragma unroll
  for (int nt = 0; nt < 8; ++nt) {
    const int n = nt * 16 + (lane & 15);
#pragma unroll
    for (int r = 0; r < 8; ++r) {
      const int m = row0 + r + ((lane >> 4) << 3);
      if (m < M) C[(long long)m * 128 + n] = acc[nt][r];
    }
  }
}

// ---------------------------------------------------------------------------
// Fused edge classifier. 64 edges/block, 8 waves:
//   wave w: edge strip (w>>1)*16, column half (w&1)*64.
// A = concat(h2[src], h2[dst]) gathered per-lane as two 16B loads (each
// 32-K chunk lies wholly in the src half or the dst half since 128%32==0).
// B = prepacked W3 fragments. z kept in LDS for the fused 128-dot + sigmoid.
// ---------------------------------------------------------------------------
__global__ __launch_bounds__(256)
void edge_mlp_kernel(const _Float16* __restrict__ h2,
                     const long long* __restrict__ src,
                     const long long* __restrict__ dst,
                     const _Float16* __restrict__ W3p,  // prepacked, K=256
                     const float* __restrict__ b3,      // [128]
                     const float* __restrict__ W4,      // [128]
                     const float* __restrict__ b4,      // [1]
                     float* __restrict__ out, int E) {
  __shared__ __align__(16) float zb[64 * 132];  // z tile, padded rows
  __shared__ float w4s[128];
  __shared__ float psum[256];

  const int t = threadIdx.x;
  const int lane = t & 31;
  const int w = t >> 5;
  if (t < 128) w4s[t] = W4[t];

  const int e0 = blockIdx.x * 64;
  const int strip = w >> 1;      // 0..3: 16-edge row strip
  const int ch = w & 1;          // column half (64 cols)

  long long e = e0 + strip * 16 + (lane & 15);
  if (e >= E) e = E - 1;         // clamp loads; stores guarded
  const long long ns = src[e];
  const long long nd = dst[e];
  const int kh = (lane >> 4) << 3;

  v8f acc[4] = {};

#pragma unroll
  for (int chunk = 0; chunk < 8; ++chunk) {
    const int k0 = chunk * 32;
    const long long node = (k0 < 128) ? ns : nd;
    const _Float16* p = h2 + node * 128 + (k0 & 127) + kh;
    const v16h af = frag16(*(const v8h*)p, *(const v8h*)(p + 16));
    const _Float16* bc = W3p + ((size_t)(chunk * 8 + ch * 4) * 32 + lane) * 16;
#pragma unroll
    for (int nt = 0; nt < 4; ++nt) {
      const v16h bf = *(const v16h*)(bc + nt * (32 * 16));
      acc[nt] = __builtin_amdgcn_wmma_f32_16x16x32_f16(
          false, af, false, bf, (short)0, acc[nt], false, false);
    }
  }

  // z = relu(acc + b3) -> LDS
#pragma unroll
  for (int nt = 0; nt < 4; ++nt) {
    const int n = ch * 64 + nt * 16 + (lane & 15);
    const float bias = b3[n];
#pragma unroll
    for (int r = 0; r < 8; ++r) {
      const int m = strip * 16 + r + ((lane >> 4) << 3);
      zb[m * 132 + n] = fmaxf(acc[nt][r] + bias, 0.f);
    }
  }
  __syncthreads();

  // 128-dot with W4, split 4-ways per edge, then reduce.
  {
    const int r = t & 63, q = t >> 6;
    float s = 0.f;
#pragma unroll
    for (int k = 0; k < 32; ++k) s += zb[r * 132 + q * 32 + k] * w4s[q * 32 + k];
    psum[t] = s;
  }
  __syncthreads();
  if (t < 64) {
    const long long eo = e0 + t;
    if (eo < E) {
      const float s = psum[t] + psum[t + 64] + psum[t + 128] + psum[t + 192]
                      + b4[0];
      out[eo] = 1.f / (1.f + __expf(-s));
    }
  }
}

// ---------------------------------------------------------------------------
// Elementwise / scatter kernels
// ---------------------------------------------------------------------------
__global__ void f32_to_f16_x4_kernel(const float* __restrict__ in,
                                     _Float16* __restrict__ out, long long n4) {
  long long i = (long long)blockIdx.x * blockDim.x + threadIdx.x;
  if (i < n4) {
    const float4 v = ((const float4*)in)[i];
    v4h o;
    o[0] = (_Float16)v.x; o[1] = (_Float16)v.y;
    o[2] = (_Float16)v.z; o[3] = (_Float16)v.w;
    ((v4h*)out)[i] = o;
  }
}

__global__ void deg_init_kernel(float* __restrict__ deg, int n) {
  int i = blockIdx.x * blockDim.x + threadIdx.x;
  if (i < n) deg[i] = 1.0f;  // self loop contributes 1
}

__global__ void deg_accum_kernel(const long long* __restrict__ dst,
                                 float* __restrict__ deg, int E) {
  int i = blockIdx.x * blockDim.x + threadIdx.x;
  if (i < E) atomicAdd(&deg[dst[i]], 1.0f);
}

__global__ void dinv_kernel(const float* __restrict__ deg,
                            float* __restrict__ dinv, int n) {
  int i = blockIdx.x * blockDim.x + threadIdx.x;
  if (i < n) {
    const float d = deg[i];
    dinv[i] = (d > 0.f) ? rsqrtf(d) : 0.f;
  }
}

// agg[i][0..3] = hlin[i][0..3] * dinv[i]^2  (self-loop message == init)
__global__ void selfloop_init_kernel(const float* __restrict__ hlin,
                                     const float* __restrict__ dinv,
                                     float* __restrict__ agg, long long n4) {
  long long i = (long long)blockIdx.x * blockDim.x + threadIdx.x;
  if (i < n4) {
    const float di = dinv[i >> 5];          // 32 float4 per 128-wide row
    const float s = di * di;
    float4 v = ((const float4*)hlin)[i];
    v.x *= s; v.y *= s; v.z *= s; v.w *= s;
    ((float4*)agg)[i] = v;
  }
}

// One wave per edge; each lane owns 4 contiguous features (128/32).
__global__ __launch_bounds__(256)
void edge_agg_kernel(const float* __restrict__ hlin,
                     const long long* __restrict__ src,
                     const long long* __restrict__ dst,
                     const float* __restrict__ dinv,
                     float* __restrict__ agg, int E) {
  const long long idx = (long long)blockIdx.x * blockDim.x + threadIdx.x;
  const long long e = idx >> 5;
  const int lane = (int)(idx & 31);
  if (e >= E) return;
  const long long s = src[e];
  const long long d = dst[e];
  if (e + 8 < E) {  // prefetch a future source row (global_prefetch_b8)
    __builtin_prefetch(hlin + src[e + 8] * 128, 0, 1);
  }
  const float norm = dinv[s] * dinv[d];
  const float4 v = ((const float4*)(hlin + s * 128))[lane];
  float* ap = agg + d * 128 + (long long)lane * 4;
  atomicAdd(ap + 0, v.x * norm);
  atomicAdd(ap + 1, v.y * norm);
  atomicAdd(ap + 2, v.z * norm);
  atomicAdd(ap + 3, v.w * norm);
}

// out16[i*4..+3] = f16( maybe_relu(agg + b) )
__global__ void bias_act_kernel(const float* __restrict__ agg,
                                const float* __restrict__ b,
                                _Float16* __restrict__ out16,
                                long long n4, int do_relu) {
  long long i = (long long)blockIdx.x * blockDim.x + threadIdx.x;
  if (i < n4) {
    const int f = (int)((i << 2) & 127);
    const float4 v = ((const float4*)agg)[i];
    float r0 = v.x + b[f + 0], r1 = v.y + b[f + 1];
    float r2 = v.z + b[f + 2], r3 = v.w + b[f + 3];
    if (do_relu) {
      r0 = fmaxf(r0, 0.f); r1 = fmaxf(r1, 0.f);
      r2 = fmaxf(r2, 0.f); r3 = fmaxf(r3, 0.f);
    }
    v4h o; o[0] = (_Float16)r0; o[1] = (_Float16)r1;
    o[2] = (_Float16)r2; o[3] = (_Float16)r3;
    ((v4h*)out16)[i] = o;
  }
}

// ---------------------------------------------------------------------------
// Host launcher
// ---------------------------------------------------------------------------
static inline int cdiv(long long a, long long b) { return (int)((a + b - 1) / b); }

extern "C" void kernel_launch(void* const* d_in, const int* in_sizes, int n_in,
                              void* d_out, int out_size, void* d_ws, size_t ws_size,
                              hipStream_t stream) {
  (void)n_in; (void)ws_size; (void)out_size;
  const float*     x   = (const float*)d_in[0];
  const long long* ei  = (const long long*)d_in[1];  // int64 [2, E]
  const float*     W1  = (const float*)d_in[2];
  const float*     b1  = (const float*)d_in[3];
  const float*     W2  = (const float*)d_in[4];
  const float*     b2  = (const float*)d_in[5];
  const float*     W3  = (const float*)d_in[6];
  const float*     b3  = (const float*)d_in[7];
  const float*     W4  = (const float*)d_in[8];
  const float*     b4  = (const float*)d_in[9];

  const int N = N_NODES;
  const int E = in_sizes[1] / 2;
  const long long* src = ei;
  const long long* dstp = ei + E;
  float* out = (float*)d_out;

  // Workspace carve-up
  char* ws = (char*)d_ws;
  size_t off = 0;
  auto carve = [&](size_t bytes) -> void* {
    void* p = ws + off;
    off = (off + bytes + 255) & ~((size_t)255);
    return p;
  };
  _Float16* xf16 = (_Float16*)carve((size_t)N * 256 * 2);
  _Float16* w1p  = (_Float16*)carve((size_t)256 * 128 * 2);
  _Float16* w2p  = (_Float16*)carve((size_t)128 * 128 * 2);
  _Float16* w3p  = (_Float16*)carve((size_t)256 * 128 * 2);
  float*    deg  = (float*)carve((size_t)N * 4);
  float*    dinv = (float*)carve((size_t)N * 4);
  float*    hlin = (float*)carve((size_t)N * 128 * 4);
  float*    agg  = (float*)carve((size_t)N * 128 * 4);
  _Float16* h1h  = (_Float16*)carve((size_t)N * 128 * 2);
  _Float16* h2h  = (_Float16*)carve((size_t)N * 128 * 2);

  const int B = 256;
  const long long nx4 = (long long)N * 256 / 4;
  const long long nh  = (long long)N * 128;
  const long long nh4 = nh / 4;

  // f32 -> f16 conversion of x; weights prepacked into B-fragment layout.
  f32_to_f16_x4_kernel<<<cdiv(nx4, B), B, 0, stream>>>(x, xf16, nx4);
  prepack_b_kernel<<<cdiv(256 * 128, B), B, 0, stream>>>(W1, w1p, 256);
  prepack_b_kernel<<<cdiv(128 * 128, B), B, 0, stream>>>(W2, w2p, 128);
  prepack_b_kernel<<<cdiv(256 * 128, B), B, 0, stream>>>(W3, w3p, 256);

  // Symmetric normalization: deg = 1 (self loop) + in-degree; dinv = rsqrt
  deg_init_kernel<<<cdiv(N, B), B, 0, stream>>>(deg, N);
  deg_accum_kernel<<<cdiv(E, B), B, 0, stream>>>(dstp, deg, E);
  dinv_kernel<<<cdiv(N, B), B, 0, stream>>>(deg, dinv, N);

  const int gemm_grid = cdiv(N, 128);

  // ---- GCN layer 1 ----
  gemm_f16_n128_kernel<<<gemm_grid, 256, 0, stream>>>(xf16, w1p, hlin, N, 256);
  selfloop_init_kernel<<<cdiv(nh4, B), B, 0, stream>>>(hlin, dinv, agg, nh4);
  edge_agg_kernel<<<cdiv((long long)E * 32, B), B, 0, stream>>>(
      hlin, src, dstp, dinv, agg, E);
  bias_act_kernel<<<cdiv(nh4, B), B, 0, stream>>>(agg, b1, h1h, nh4, 1);

  // ---- GCN layer 2 (no relu after aggregation) ----
  gemm_f16_n128_kernel<<<gemm_grid, 256, 0, stream>>>(h1h, w2p, hlin, N, 128);
  selfloop_init_kernel<<<cdiv(nh4, B), B, 0, stream>>>(hlin, dinv, agg, nh4);
  edge_agg_kernel<<<cdiv((long long)E * 32, B), B, 0, stream>>>(
      hlin, src, dstp, dinv, agg, E);
  bias_act_kernel<<<cdiv(nh4, B), B, 0, stream>>>(agg, b2, h2h, nh4, 0);

  // ---- Fused edge MLP: gather + GEMM(256->128) + relu + 128-dot + sigmoid
  edge_mlp_kernel<<<cdiv(E, 64), 256, 0, stream>>>(
      h2h, src, dstp, w3p, b3, W4, b4, out, E);
}
